// Attention_13142599926197
// MI455X (gfx1250) — matrix operand order repaired and checked by reading, not compiled, not run on hardware
//
#include <hip/hip_runtime.h>
#include <hip/hip_bf16.h>
#include <math.h>

typedef __attribute__((ext_vector_type(16))) __bf16 v16bf;
typedef __attribute__((ext_vector_type(8)))  float  v8f;

constexpr int B  = 64;
constexpr int Te = 1024;
constexpr int De = 1024;
constexpr int Dd = 1024;
constexpr int H  = 1024;

constexpr int HC  = 128;            // h-rows per LDS slab (= full N range per WG pass)
constexpr int KC  = 64;             // k-depth per LDS slab
constexpr int PAD = 8;              // row padding (bf16 elems) to spread LDS banks
constexpr int NS  = (H / HC) * (De / KC);   // 128 slabs

// ---------------------------------------------------------------------------
// Kernel 1: Wa_vec[b,h] = dot(dec[b,:], Wa_w[h,:]) + Wa_b[h] + Ua_b[h]
// ---------------------------------------------------------------------------
__global__ __launch_bounds__(256) void wa_kernel(
    const float* __restrict__ dec, const float* __restrict__ Wa_w,
    const float* __restrict__ Wa_b, const float* __restrict__ Ua_b,
    float* __restrict__ wa_vec) {
  const int idx = blockIdx.x * blockDim.x + threadIdx.x;   // [0, B*H)
  const int b = idx >> 10;
  const int h = idx & (H - 1);
  const float* d = dec  + (size_t)b * Dd;
  const float* w = Wa_w + (size_t)h * Dd;
  float acc = 0.f;
  for (int k = 0; k < Dd; k += 4) {
    float4 dv = *(const float4*)(d + k);
    float4 wv = *(const float4*)(w + k);
    acc = fmaf(dv.x, wv.x, acc);
    acc = fmaf(dv.y, wv.y, acc);
    acc = fmaf(dv.z, wv.z, acc);
    acc = fmaf(dv.w, wv.w, acc);
  }
  wa_vec[idx] = acc + Wa_b[h] + Ua_b[h];
}

// ---------------------------------------------------------------------------
// Kernel 2 (dominant): fused  Ua-GEMM -> +Wa_vec -> tanh -> dot(Va_w) -> score
//
// Grid: (Te/128, B); block = 256 = 8 waves. Wave w owns e-rows
// [eBase + 16w, +16) and sweeps the FULL H, so the score reduction finishes
// inside the wave. Ua_w is staged through LDS as bf16 (converted once,
// shared by all 8 waves), double-buffered 128x64 slabs.
//
// Wave32 bf16 fragment layouts (ISA 7.12.2):
//  A 16x32: lane m = lane&15; lanes<16 hold K {0..7,16..23}, lanes>=16 hold
//           K {8..15,24..31}.
//  B 32x16: lane n = lane&15; lanes<16 hold K {0..7,16..23}, lanes>=16 K {8..15,24..31}.
//  D 16x16: value(vgpr v, lane l) = D[m = v + (l>=16)*8, n = l&15].
// ---------------------------------------------------------------------------
__global__ __launch_bounds__(256) void score_kernel(
    const float* __restrict__ enc, const float* __restrict__ Ua_w,
    const float* __restrict__ wa_vec, const float* __restrict__ Va_w,
    float* __restrict__ scores) {
  const int eBase = blockIdx.x * 128;
  const int b     = blockIdx.y;
  const int tid   = threadIdx.x;
  const int wave  = tid >> 5;
  const int lane  = tid & 31;
  const int lhalf = lane >> 4;
  const int mn    = lane & 15;

  __shared__ __align__(16) __bf16 Bs[2][HC][KC + PAD];   // 2 x 18 KB

  // staging role: thread -> (row sr, 32-col chunk scB) of the 128x64 slab
  const int sr  = tid >> 1;
  const int scB = (tid & 1) * 32;

  // A row pointer for this lane (row e = eBase + wave*16 + mn)
  const float* arow =
      enc + ((size_t)b * Te + (eBase + wave * 16 + mn)) * De + lhalf * 8;

  float psum[8];
#pragma unroll
  for (int v = 0; v < 8; ++v) psum[v] = 0.f;

  v8f acc[8];

  float4 sreg[8];
  // ---- prologue: stage slab 0 ----
  {
    const float* g = Ua_w + (size_t)sr * De + scB;
#pragma unroll
    for (int i = 0; i < 8; ++i) sreg[i] = *(const float4*)(g + i * 4);
    union { __bf16 h[32]; uint4 q[4]; } up;
#pragma unroll
    for (int i = 0; i < 8; ++i) {
      up.h[i * 4 + 0] = (__bf16)sreg[i].x; up.h[i * 4 + 1] = (__bf16)sreg[i].y;
      up.h[i * 4 + 2] = (__bf16)sreg[i].z; up.h[i * 4 + 3] = (__bf16)sreg[i].w;
    }
    uint4* p = (uint4*)&Bs[0][sr][scB];
#pragma unroll
    for (int i = 0; i < 4; ++i) p[i] = up.q[i];
  }
  __syncthreads();

  for (int s = 0; s < NS; ++s) {
    const int hcs = s >> 4;              // h-chunk index (0..7)
    const int kc  = (s & 15) * KC;       // k-base of this slab
    const int buf = s & 1;
    const bool hasNext = (s + 1 < NS);

    // issue next slab's global loads early (latency overlapped with WMMAs)
    if (hasNext) {
      const int sn  = s + 1;
      const int hcn = sn >> 4;
      const int kcn = (sn & 15) * KC;
      const float* g = Ua_w + (size_t)(hcn * HC + sr) * De + kcn + scB;
#pragma unroll
      for (int i = 0; i < 8; ++i) sreg[i] = *(const float4*)(g + i * 4);
    }

    if (kc == 0) {
#pragma unroll
      for (int t = 0; t < 8; ++t)
        acc[t] = (v8f){0.f, 0.f, 0.f, 0.f, 0.f, 0.f, 0.f, 0.f};
    }

    // ---- 2 WMMA k-substeps over this slab ----
#pragma unroll
    for (int ks = 0; ks < KC; ks += 32) {
      const float* ap = arow + kc + ks;
      float4 a0 = *(const float4*)(ap);
      float4 a1 = *(const float4*)(ap + 4);
      float4 a2 = *(const float4*)(ap + 16);
      float4 a3 = *(const float4*)(ap + 20);
      v16bf af;
      af[0]  = (__bf16)a0.x; af[1]  = (__bf16)a0.y; af[2]  = (__bf16)a0.z; af[3]  = (__bf16)a0.w;
      af[4]  = (__bf16)a1.x; af[5]  = (__bf16)a1.y; af[6]  = (__bf16)a1.z; af[7]  = (__bf16)a1.w;
      af[8]  = (__bf16)a2.x; af[9]  = (__bf16)a2.y; af[10] = (__bf16)a2.z; af[11] = (__bf16)a2.w;
      af[12] = (__bf16)a3.x; af[13] = (__bf16)a3.y; af[14] = (__bf16)a3.z; af[15] = (__bf16)a3.w;

      const int dOff = ks + lhalf * 8;
#pragma unroll
      for (int t = 0; t < 8; ++t) {
        const __bf16* bp = &Bs[buf][t * 16 + mn][dOff];
        union { uint4 q[2]; v16bf v; } ub;
        ub.q[0] = *(const uint4*)bp;
        ub.q[1] = *(const uint4*)(bp + 16);
        acc[t] = __builtin_amdgcn_wmma_f32_16x16x32_bf16(
            /*neg_a=*/false, af, /*neg_b=*/false, ub.v,
            /*c_mod=*/(short)0, acc[t], /*reuse_a=*/false, /*reuse_b=*/false);
      }
    }

    // ---- epilogue for a finished h-chunk: tanh(Ua+Wa)*Va_w into psum ----
    if (kc == De - KC) {
#pragma unroll
      for (int t = 0; t < 8; ++t) {
        const int h = hcs * HC + t * 16 + mn;   // N index of this lane
        const float wav = wa_vec[(size_t)b * H + h];
        const float vaw = Va_w[h];
#pragma unroll
        for (int v = 0; v < 8; ++v) {
          float x = tanhf(acc[t][v] + wav);
          psum[v] = fmaf(x, vaw, psum[v]);      // partial for row m = v + lhalf*8
        }
      }
    }

    // ---- store next slab into the other LDS buffer ----
    if (hasNext) {
      union { __bf16 h[32]; uint4 q[4]; } up;
#pragma unroll
      for (int i = 0; i < 8; ++i) {
        up.h[i * 4 + 0] = (__bf16)sreg[i].x; up.h[i * 4 + 1] = (__bf16)sreg[i].y;
        up.h[i * 4 + 2] = (__bf16)sreg[i].z; up.h[i * 4 + 3] = (__bf16)sreg[i].w;
      }
      uint4* p = (uint4*)&Bs[buf ^ 1][sr][scB];
#pragma unroll
      for (int i = 0; i < 4; ++i) p[i] = up.q[i];
    }
    __syncthreads();
  }

  // ---- reduce over the 16 N-lanes within each half; write 16 scores ----
#pragma unroll
  for (int v = 0; v < 8; ++v) {
    float sv = psum[v];
    sv += __shfl_xor(sv, 1);
    sv += __shfl_xor(sv, 2);
    sv += __shfl_xor(sv, 4);
    sv += __shfl_xor(sv, 8);
    if (mn == 0) {
      const int e = eBase + wave * 16 + v + lhalf * 8;
      // Va_b omitted: uniform shift, cancels in softmax.
      scores[(size_t)b * Te + e] = sv;
    }
  }
}

// ---------------------------------------------------------------------------
// Kernel 3: in-place softmax over Te per batch row
// ---------------------------------------------------------------------------
__global__ __launch_bounds__(256) void softmax_kernel(float* __restrict__ scores) {
  const int b = blockIdx.x;
  const int tid = threadIdx.x;
  __shared__ float red[256];
  float* row = scores + (size_t)b * Te;

  float x[4];
  float m = -INFINITY;
#pragma unroll
  for (int j = 0; j < 4; ++j) { x[j] = row[tid + j * 256]; m = fmaxf(m, x[j]); }
  red[tid] = m; __syncthreads();
  for (int s = 128; s > 0; s >>= 1) {
    if (tid < s) red[tid] = fmaxf(red[tid], red[tid + s]);
    __syncthreads();
  }
  m = red[0]; __syncthreads();

  float sum = 0.f;
#pragma unroll
  for (int j = 0; j < 4; ++j) { x[j] = __expf(x[j] - m); sum += x[j]; }
  red[tid] = sum; __syncthreads();
  for (int s = 128; s > 0; s >>= 1) {
    if (tid < s) red[tid] += red[tid + s];
    __syncthreads();
  }
  const float inv = 1.f / red[0];
#pragma unroll
  for (int j = 0; j < 4; ++j) row[tid + j * 256] = x[j] * inv;
}

// ---------------------------------------------------------------------------
// Kernel 4: context[b,d] = sum_e weights[b,e] * enc[b,e,d]
// ---------------------------------------------------------------------------
__global__ __launch_bounds__(256) void context_kernel(
    const float* __restrict__ enc, const float* __restrict__ weights,
    float* __restrict__ out) {
  const int b = blockIdx.x;
  const int tid = threadIdx.x;
  __shared__ float w[Te];
  for (int e = tid; e < Te; e += 256) w[e] = weights[(size_t)b * Te + e];
  __syncthreads();

  float acc0 = 0.f, acc1 = 0.f, acc2 = 0.f, acc3 = 0.f;
  const float* eb = enc + (size_t)b * Te * De;
  for (int e = 0; e < Te; ++e) {
    const float we = w[e];
    const float* r = eb + (size_t)e * De;
    acc0 = fmaf(we, r[tid], acc0);
    acc1 = fmaf(we, r[tid + 256], acc1);
    acc2 = fmaf(we, r[tid + 512], acc2);
    acc3 = fmaf(we, r[tid + 768], acc3);
  }
  float* o = out + (size_t)b * De;
  o[tid]       = acc0;
  o[tid + 256] = acc1;
  o[tid + 512] = acc2;
  o[tid + 768] = acc3;
}

// ---------------------------------------------------------------------------
extern "C" void kernel_launch(void* const* d_in, const int* in_sizes, int n_in,
                              void* d_out, int out_size, void* d_ws, size_t ws_size,
                              hipStream_t stream) {
  (void)in_sizes; (void)n_in; (void)out_size; (void)ws_size;
  const float* enc  = (const float*)d_in[0];
  const float* dec  = (const float*)d_in[1];
  const float* Wa_w = (const float*)d_in[2];
  const float* Wa_b = (const float*)d_in[3];
  const float* Ua_w = (const float*)d_in[4];
  const float* Ua_b = (const float*)d_in[5];
  const float* Va_w = (const float*)d_in[6];
  // d_in[7] = Va_b: uniform score shift, softmax-invariant -> unused.
  float* out    = (float*)d_out;
  float* wa_vec = (float*)d_ws;            // B*H floats (256 KB)
  float* scores = wa_vec + (size_t)B * H;  // B*Te floats (256 KB)

  wa_kernel<<<(B * H) / 256, 256, 0, stream>>>(dec, Wa_w, Wa_b, Ua_b, wa_vec);

  dim3 g2(Te / 128, B);
  score_kernel<<<g2, 256, 0, stream>>>(enc, Ua_w, wa_vec, Va_w, scores);

  softmax_kernel<<<B, 256, 0, stream>>>(scores);

  context_kernel<<<B, 256, 0, stream>>>(enc, scores, out);
}
